// Model_20650202759740
// MI455X (gfx1250) — matrix operand order repaired and checked by reading, not compiled, not run on hardware
//
#include <hip/hip_runtime.h>

// ---------------------------------------------------------------------------
// Types for CDNA5 WMMA (wave32)
// ---------------------------------------------------------------------------
typedef __attribute__((ext_vector_type(16))) _Float16 v16h;
typedef __attribute__((ext_vector_type(8)))  _Float16 v8h;
typedef __attribute__((ext_vector_type(2)))  _Float16 v2h;
typedef __attribute__((ext_vector_type(8)))  float    v8f;

// Model dims
#define BB 32
#define NN 256
#define TT 64
#define FF 16
#define HH 128
#define GATES 512         // 4*H
#define ROWS 32           // sequences per block
#define BLOCK 256         // 8 waves
#define WHH_HALVES (32 * 4 * 32 * 16)   // 32 col-tiles * 4 k-tiles * 512 halves

// ---------------------------------------------------------------------------
// Device helpers
// ---------------------------------------------------------------------------
__device__ __forceinline__ float sig_f(float x)  { return 1.0f / (1.0f + __expf(-x)); }
__device__ __forceinline__ float tanh_f(float x) { return 1.0f - 2.0f / (__expf(2.0f * x) + 1.0f); }

// A-operand gather from LDS (row-major tile, `stride` halves per row) into the
// ISA 16-bit A layout: lane L -> row = L&15, kbase = (L>>4)*8,
// elements 0..7 -> K = kt*32+kbase+0..7, elements 8..15 -> K = kt*32+kbase+16..23
__device__ __forceinline__ v16h load_a_lds(const _Float16* __restrict__ sh,
                                           int stride, int kt, int lane) {
  const int row = lane & 15;
  const int kb  = (lane >> 4) * 8;
  const _Float16* p = sh + row * stride + kt * 32 + kb;
  v8h lo = *(const v8h*)(p);
  v8h hi = *(const v8h*)(p + 16);
  v16h a;
#pragma unroll
  for (int i = 0; i < 8; ++i) { a[i] = lo[i]; a[i + 8] = hi[i]; }
  return a;
}

// Pre-swizzled B tiles: [tile][lane][16 halves] -> one contiguous 32B read/lane
__device__ __forceinline__ v16h load_b_gl(const _Float16* __restrict__ w, int tile, int lane) {
  return *(const v16h*)(w + ((size_t)tile * 32 + lane) * 16);
}
__device__ __forceinline__ v16h load_b_lds(const _Float16* sh, int tile, int lane) {
  return *(const v16h*)(sh + ((size_t)tile * 32 + lane) * 16);
}

// In-register LSTM cell update for one wave's 16 h-columns, 32 rows.
// acc[G][rt] holds gate G pre-activation (G: 0=i,1=f,2=g,3=o).
__device__ __forceinline__ void cell_update_reg(v8f acc[4][2],
                                                float bi, float bf, float bg, float bo,
                                                float cst[2][8],
                                                _Float16* sh_h, int colw, int rb) {
#pragma unroll
  for (int rt = 0; rt < 2; ++rt) {
#pragma unroll
    for (int r = 0; r < 8; ++r) {
      float gi = acc[0][rt][r] + bi;
      float gf = acc[1][rt][r] + bf;
      float gg = acc[2][rt][r] + bg;
      float go = acc[3][rt][r] + bo;
      float c  = sig_f(gf) * cst[rt][r] + sig_f(gi) * tanh_f(gg);
      cst[rt][r] = c;
      sh_h[(rt * 16 + rb + r) * HH + colw] = (_Float16)(sig_f(go) * tanh_f(c));
    }
  }
}

// ---------------------------------------------------------------------------
// Prep kernels
// ---------------------------------------------------------------------------
// Swizzle W (512 x K_in row-major) into WMMA B-tile lane layout (f16, K padded
// to ktiles*32):  i = ((ntile*ktiles + kt)*32 + lane)*16 + e
//                 n = ntile*16 + (lane&15);  k = kt*32 + (e>>3)*16 + (lane>>4)*8 + (e&7)
__global__ void swizzle_w(const float* __restrict__ W, _Float16* __restrict__ out,
                          int K_in, int ktiles) {
  int i = blockIdx.x * blockDim.x + threadIdx.x;
  int total = 32 * ktiles * 32 * 16;
  if (i >= total) return;
  int e    = i & 15;
  int lane = (i >> 4) & 31;
  int t9   = i >> 9;
  int kt   = t9 % ktiles;
  int nt   = t9 / ktiles;
  int n = nt * 16 + (lane & 15);
  int k = kt * 32 + (e >> 3) * 16 + (lane >> 4) * 8 + (e & 7);
  float v = (k < K_in) ? W[n * K_in + k] : 0.0f;
  out[i] = (_Float16)v;
}

__global__ void bias_comb(const float* __restrict__ bi, const float* __restrict__ bh,
                          float* __restrict__ o) {
  int i = blockIdx.x * blockDim.x + threadIdx.x;
  if (i < GATES) o[i] = bi[i] + bh[i];
}

__global__ void sigmoid_k(const float* __restrict__ a, float* __restrict__ o, int n) {
  int i = blockIdx.x * blockDim.x + threadIdx.x;
  if (i < n) o[i] = sig_f(a[i]);
}

// ---------------------------------------------------------------------------
// Layer 0 LSTM: 32 sequences/block, Whh0 LDS-resident, Wih0 in registers,
// cell state in registers, x staging software-pipelined (load t+1 during t).
// ---------------------------------------------------------------------------
__global__ __launch_bounds__(BLOCK)
void lstm_l0_kernel(const float* __restrict__ x,
                    const _Float16* __restrict__ Wih0s,
                    const _Float16* __restrict__ Whh0s,
                    const float* __restrict__ bias0,
                    _Float16* __restrict__ h0seq) {
  __shared__ _Float16 sh_whh[WHH_HALVES];   // 128 KB
  __shared__ _Float16 sh_x[ROWS * 32];      //   2 KB (K padded 16->32)
  __shared__ _Float16 sh_h[ROWS * HH];      //   8 KB

  const int tid  = threadIdx.x;
  const int lane = tid & 31;
  const int wave = tid >> 5;                // owns h-cols [wave*16, wave*16+16)
  const int colw = wave * 16 + (lane & 15);
  const int rb   = (lane >> 4) * 8;
  const size_t row0 = (size_t)blockIdx.x * ROWS;

  // per-thread staging ownership
  const int xr = tid >> 3;            // x row 0..31
  const int xf = (tid & 7) * 2;       // x feature pair 0,2,..,14
  const int hr0 = tid >> 4;           // h chunk rows
  const int hr1 = hr0 + 16;
  const int hc  = (tid & 15) * 8;     // h col chunk (8 halves)

  // Stage Whh0 into LDS once (v8h chunks)
  for (int i = tid; i < WHH_HALVES / 8; i += BLOCK)
    ((v8h*)sh_whh)[i] = ((const v8h*)Whh0s)[i];
  for (int i = tid; i < ROWS * HH; i += BLOCK) sh_h[i] = (_Float16)0.f;
  // zero the K-pad half of sh_x once (never rewritten)
  *(v2h*)(sh_x + xr * 32 + 16 + xf) = (v2h){(_Float16)0.f, (_Float16)0.f};
  // stage x_0
  {
    float2 v = *(const float2*)(x + ((row0 + xr) * TT + 0) * FF + xf);
    *(v2h*)(sh_x + xr * 32 + xf) = (v2h){(_Float16)v.x, (_Float16)v.y};
  }

  // Input-side weight slice in registers: 4 gate tiles, ktiles=1
  v16h wih[4];
#pragma unroll
  for (int g = 0; g < 4; ++g) wih[g] = load_b_gl(Wih0s, g * 8 + wave, lane);
  const float bi = bias0[colw], bf = bias0[HH + colw],
              bg = bias0[2 * HH + colw], bo = bias0[3 * HH + colw];

  float cst[2][8];
#pragma unroll
  for (int rt = 0; rt < 2; ++rt)
#pragma unroll
    for (int r = 0; r < 8; ++r) cst[rt][r] = 0.f;
  __syncthreads();

  for (int t = 0; t < TT; ++t) {
    // issue next-step global loads (latency hidden under WMMA)
    float2 xnext = {0.f, 0.f};
    if (t + 1 < TT)
      xnext = *(const float2*)(x + ((row0 + xr) * TT + (t + 1)) * FF + xf);
    if (t + 2 < TT)
      __builtin_prefetch(x + ((row0 + xr) * TT + (t + 2)) * FF + xf, 0, 1);

    v8f acc[4][2];
#pragma unroll
    for (int g = 0; g < 4; ++g)
#pragma unroll
      for (int rt = 0; rt < 2; ++rt)
        acc[g][rt] = (v8f){0.f, 0.f, 0.f, 0.f, 0.f, 0.f, 0.f, 0.f};

    {  // x @ Wih0^T  (single padded k-tile, B in registers)
      v16h a0 = load_a_lds(sh_x, 32, 0, lane);
      v16h a1 = load_a_lds(sh_x + 16 * 32, 32, 0, lane);
#pragma unroll
      for (int g = 0; g < 4; ++g) {
        acc[g][0] = __builtin_amdgcn_wmma_f32_16x16x32_f16(false, a0, false, wih[g], (short)0, acc[g][0], false, false);
        acc[g][1] = __builtin_amdgcn_wmma_f32_16x16x32_f16(false, a1, false, wih[g], (short)0, acc[g][1], false, false);
      }
    }
#pragma unroll
    for (int kt = 0; kt < 4; ++kt) {  // h @ Whh0^T  (B from LDS)
      v16h a0 = load_a_lds(sh_h, HH, kt, lane);
      v16h a1 = load_a_lds(sh_h + 16 * HH, HH, kt, lane);
#pragma unroll
      for (int g = 0; g < 4; ++g) {
        v16h b = load_b_lds(sh_whh, (g * 8 + wave) * 4 + kt, lane);
        acc[g][0] = __builtin_amdgcn_wmma_f32_16x16x32_f16(false, a0, false, b, (short)0, acc[g][0], false, false);
        acc[g][1] = __builtin_amdgcn_wmma_f32_16x16x32_f16(false, a1, false, b, (short)0, acc[g][1], false, false);
      }
    }
    __syncthreads();                       // all reads of sh_x/sh_h done
    cell_update_reg(acc, bi, bf, bg, bo, cst, sh_h, colw, rb);
    if (t + 1 < TT) {                      // commit pipelined x_{t+1}
      // scheduling fence: make this the first use of xnext so the load's
      // waitcnt (and the f32->f16 converts) sink below the WMMA block
      asm volatile("" : "+v"(xnext.x), "+v"(xnext.y));
      *(v2h*)(sh_x + xr * 32 + xf) = (v2h){(_Float16)xnext.x, (_Float16)xnext.y};
    }
    __syncthreads();                       // sh_h/sh_x fully updated

    // export h0_t to global: h0seq[row][t][h]  (two 16B chunks per thread)
    *(v8h*)(h0seq + ((row0 + hr0) * TT + t) * HH + hc) = *(const v8h*)(sh_h + hr0 * HH + hc);
    *(v8h*)(h0seq + ((row0 + hr1) * TT + t) * HH + hc) = *(const v8h*)(sh_h + hr1 * HH + hc);
  }
}

// ---------------------------------------------------------------------------
// Layer 1 LSTM: input = h0seq (f16), staging software-pipelined.
// Whh1 LDS-resident, Wih1 in registers. Writes hL[(b*128+h)*256 + node].
// ---------------------------------------------------------------------------
__global__ __launch_bounds__(BLOCK)
void lstm_l1_kernel(const _Float16* __restrict__ h0seq,
                    const _Float16* __restrict__ Wih1s,
                    const _Float16* __restrict__ Whh1s,
                    const float* __restrict__ bias1,
                    float* __restrict__ hL) {
  __shared__ _Float16 sh_whh[WHH_HALVES];   // 128 KB
  __shared__ _Float16 sh_a[ROWS * HH];      //   8 KB (staged h0_t)
  __shared__ _Float16 sh_h[ROWS * HH];      //   8 KB

  const int tid  = threadIdx.x;
  const int lane = tid & 31;
  const int wave = tid >> 5;
  const int colw = wave * 16 + (lane & 15);
  const int rb   = (lane >> 4) * 8;
  const size_t row0 = (size_t)blockIdx.x * ROWS;

  const int hr0 = tid >> 4;
  const int hr1 = hr0 + 16;
  const int hc  = (tid & 15) * 8;

  for (int i = tid; i < WHH_HALVES / 8; i += BLOCK)
    ((v8h*)sh_whh)[i] = ((const v8h*)Whh1s)[i];
  for (int i = tid; i < ROWS * HH; i += BLOCK) sh_h[i] = (_Float16)0.f;
  // stage h0_0
  *(v8h*)(sh_a + hr0 * HH + hc) = *(const v8h*)(h0seq + ((row0 + hr0) * TT + 0) * HH + hc);
  *(v8h*)(sh_a + hr1 * HH + hc) = *(const v8h*)(h0seq + ((row0 + hr1) * TT + 0) * HH + hc);

  // Input-side weight slice in registers: 4 gate tiles x 4 k-tiles
  v16h wih[4][4];
#pragma unroll
  for (int g = 0; g < 4; ++g)
#pragma unroll
    for (int kt = 0; kt < 4; ++kt)
      wih[g][kt] = load_b_gl(Wih1s, (g * 8 + wave) * 4 + kt, lane);
  const float bi = bias1[colw], bf = bias1[HH + colw],
              bg = bias1[2 * HH + colw], bo = bias1[3 * HH + colw];

  float cst[2][8];
#pragma unroll
  for (int rt = 0; rt < 2; ++rt)
#pragma unroll
    for (int r = 0; r < 8; ++r) cst[rt][r] = 0.f;
  __syncthreads();

  for (int t = 0; t < TT; ++t) {
    // issue next-step global loads (latency hidden under WMMA)
    v8h n0 = {(_Float16)0.f, (_Float16)0.f, (_Float16)0.f, (_Float16)0.f,
              (_Float16)0.f, (_Float16)0.f, (_Float16)0.f, (_Float16)0.f};
    v8h n1 = n0;
    if (t + 1 < TT) {
      n0 = *(const v8h*)(h0seq + ((row0 + hr0) * TT + (t + 1)) * HH + hc);
      n1 = *(const v8h*)(h0seq + ((row0 + hr1) * TT + (t + 1)) * HH + hc);
    }
    if (t + 2 < TT)
      __builtin_prefetch(h0seq + ((row0 + hr0) * TT + (t + 2)) * HH + hc, 0, 1);

    v8f acc[4][2];
#pragma unroll
    for (int g = 0; g < 4; ++g)
#pragma unroll
      for (int rt = 0; rt < 2; ++rt)
        acc[g][rt] = (v8f){0.f, 0.f, 0.f, 0.f, 0.f, 0.f, 0.f, 0.f};

#pragma unroll
    for (int kt = 0; kt < 4; ++kt) {  // h0_t @ Wih1^T (B in registers)
      v16h a0 = load_a_lds(sh_a, HH, kt, lane);
      v16h a1 = load_a_lds(sh_a + 16 * HH, HH, kt, lane);
#pragma unroll
      for (int g = 0; g < 4; ++g) {
        acc[g][0] = __builtin_amdgcn_wmma_f32_16x16x32_f16(false, a0, false, wih[g][kt], (short)0, acc[g][0], false, false);
        acc[g][1] = __builtin_amdgcn_wmma_f32_16x16x32_f16(false, a1, false, wih[g][kt], (short)0, acc[g][1], false, false);
      }
    }
#pragma unroll
    for (int kt = 0; kt < 4; ++kt) {  // h1 @ Whh1^T (B from LDS)
      v16h a0 = load_a_lds(sh_h, HH, kt, lane);
      v16h a1 = load_a_lds(sh_h + 16 * HH, HH, kt, lane);
#pragma unroll
      for (int g = 0; g < 4; ++g) {
        v16h b = load_b_lds(sh_whh, (g * 8 + wave) * 4 + kt, lane);
        acc[g][0] = __builtin_amdgcn_wmma_f32_16x16x32_f16(false, a0, false, b, (short)0, acc[g][0], false, false);
        acc[g][1] = __builtin_amdgcn_wmma_f32_16x16x32_f16(false, a1, false, b, (short)0, acc[g][1], false, false);
      }
    }
    __syncthreads();                       // all reads of sh_a/sh_h done
    cell_update_reg(acc, bi, bf, bg, bo, cst, sh_h, colw, rb);
    if (t + 1 < TT) {                      // commit pipelined h0_{t+1}
      *(v8h*)(sh_a + hr0 * HH + hc) = n0;
      *(v8h*)(sh_a + hr1 * HH + hc) = n1;
    }
    __syncthreads();
  }

  // final hidden transposed: hL[(b*128 + h)*256 + node]
  for (int i = tid; i < ROWS * HH; i += BLOCK) {
    int row = i >> 7, h = i & 127;
    size_t rg = row0 + row;
    hL[((rg >> 8) * HH + h) * NN + (rg & 255)] = (float)sh_h[i];
  }
}

// ---------------------------------------------------------------------------
// Graph-part kernels (fp32, ~1% of total FLOPs)
// ---------------------------------------------------------------------------
__global__ void gemm_right_A(const float* __restrict__ X, const float* __restrict__ Ap,
                             float* __restrict__ C) {
  int i = blockIdx.x * blockDim.x + threadIdx.x;  // 32*128*256
  int n  = i & 255;
  int bh = i >> 8;
  const float* xr = X + (size_t)bh * NN;
  float s = 0.f;
  for (int l = 0; l < NN; ++l) s += xr[l] * Ap[l * NN + n];
  C[i] = s;
}

__global__ void gemm_w12(const float* __restrict__ HA, const float* __restrict__ W1,
                         const float* __restrict__ W2, float* __restrict__ M1,
                         float* __restrict__ M2) {
  int i = blockIdx.x * blockDim.x + threadIdx.x;
  int n = i & 255;
  int g = (i >> 8) & 127;
  int b = i >> 15;
  const float* ha = HA + (size_t)b * HH * NN + n;
  float s1 = 0.f, s2 = 0.f;
  for (int h = 0; h < HH; ++h) {
    float v = ha[h * NN];
    s1 += W1[g * HH + h] * v;
    s2 += W2[g * HH + h] * v;
  }
  M1[i] = s1;
  M2[i] = fmaxf(s2, 0.f);
}

__global__ void gemm_w3(const float* __restrict__ M2, const float* __restrict__ W3,
                        float* __restrict__ M3) {
  int i = blockIdx.x * blockDim.x + threadIdx.x;
  int n = i & 255;
  int f = (i >> 8) & 127;
  int b = i >> 15;
  const float* m = M2 + (size_t)b * HH * NN + n;
  float s = 0.f;
  for (int g = 0; g < HH; ++g) s += W3[f * HH + g] * m[g * NN];
  M3[i] = s;
}

__global__ void final_k(const float* __restrict__ ATW, const float* __restrict__ M1,
                        const float* __restrict__ Wfc, float* __restrict__ out) {
  int i = blockIdx.x * blockDim.x + threadIdx.x;  // 32*256
  if (i >= BB * NN) return;
  int n = i & 255, b = i >> 8;
  const float* a = ATW + (size_t)b * HH * NN + n;
  const float* m = M1 + (size_t)b * HH * NN + n;
  float s = 0.f;
  for (int h = 0; h < HH; ++h) s += Wfc[h] * (0.1f * a[h * NN] + 0.9f * m[h * NN]);
  out[i] = sig_f(s);
}

// ---------------------------------------------------------------------------
// Host launcher
// ---------------------------------------------------------------------------
extern "C" void kernel_launch(void* const* d_in, const int* in_sizes, int n_in,
                              void* d_out, int out_size, void* d_ws, size_t ws_size,
                              hipStream_t stream) {
  (void)in_sizes; (void)n_in; (void)out_size; (void)ws_size;
  const float* x     = (const float*)d_in[0];
  const float* A     = (const float*)d_in[1];
  const float* W_ih0 = (const float*)d_in[2];
  const float* W_hh0 = (const float*)d_in[3];
  const float* b_ih0 = (const float*)d_in[4];
  const float* b_hh0 = (const float*)d_in[5];
  const float* W_ih1 = (const float*)d_in[6];
  const float* W_hh1 = (const float*)d_in[7];
  const float* b_ih1 = (const float*)d_in[8];
  const float* b_hh1 = (const float*)d_in[9];
  const float* W1    = (const float*)d_in[10];
  const float* W2    = (const float*)d_in[11];
  const float* W3    = (const float*)d_in[12];
  const float* Wfc   = (const float*)d_in[13];
  float* out = (float*)d_out;

  // Workspace carve (~146 MB)
  char* base = (char*)d_ws;
  size_t off = 0;
  auto carve = [&](size_t bytes) -> char* {
    char* p = base + off;
    off = (off + bytes + 255) & ~(size_t)255;
    return p;
  };
  _Float16* h0seq = (_Float16*)carve((size_t)BB * NN * TT * HH * 2);  // 128 MB
  _Float16* Wih0s = (_Float16*)carve((size_t)32 * 1 * 32 * 16 * 2);
  _Float16* Whh0s = (_Float16*)carve((size_t)WHH_HALVES * 2);
  _Float16* Wih1s = (_Float16*)carve((size_t)WHH_HALVES * 2);
  _Float16* Whh1s = (_Float16*)carve((size_t)WHH_HALVES * 2);
  float* bias0 = (float*)carve(GATES * 4);
  float* bias1 = (float*)carve(GATES * 4);
  float* Ap    = (float*)carve((size_t)NN * NN * 4);
  const size_t big = (size_t)BB * HH * NN * 4;  // 4 MB each
  float* hL = (float*)carve(big);   // later reused as M3
  float* HA = (float*)carve(big);   // later reused as ATW
  float* M1 = (float*)carve(big);
  float* M2 = (float*)carve(big);

  // 1) prep
  swizzle_w<<<64, 256, 0, stream>>>(W_ih0, Wih0s, 16, 1);
  swizzle_w<<<256, 256, 0, stream>>>(W_hh0, Whh0s, 128, 4);
  swizzle_w<<<256, 256, 0, stream>>>(W_ih1, Wih1s, 128, 4);
  swizzle_w<<<256, 256, 0, stream>>>(W_hh1, Whh1s, 128, 4);
  bias_comb<<<2, 256, 0, stream>>>(b_ih0, b_hh0, bias0);
  bias_comb<<<2, 256, 0, stream>>>(b_ih1, b_hh1, bias1);
  sigmoid_k<<<256, 256, 0, stream>>>(A, Ap, NN * NN);

  // 2) LSTM layers (WMMA, LDS-resident recurrent weights, register cell state,
  //    pipelined staging)
  lstm_l0_kernel<<<(BB * NN) / ROWS, BLOCK, 0, stream>>>(x, Wih0s, Whh0s, bias0, h0seq);
  lstm_l1_kernel<<<(BB * NN) / ROWS, BLOCK, 0, stream>>>(h0seq, Wih1s, Whh1s, bias1, hL);

  // 3) graph part
  const int gthreads = BB * HH * NN;  // 1,048,576
  gemm_right_A<<<gthreads / 256, 256, 0, stream>>>(hL, Ap, HA);      // HA = hL @ Ap
  gemm_w12<<<gthreads / 256, 256, 0, stream>>>(HA, W1, W2, M1, M2);  // M1, M2
  gemm_w3<<<gthreads / 256, 256, 0, stream>>>(M2, W3, hL);           // M3 -> reuse hL
  gemm_right_A<<<gthreads / 256, 256, 0, stream>>>(hL, Ap, HA);      // ATW -> reuse HA
  final_k<<<(BB * NN + 255) / 256, 256, 0, stream>>>(HA, M1, Wfc, out);
}
